// ConvertedDilatedLSTM_19490561589790
// MI455X (gfx1250) — compile-verified, matrix operand order checked by reading, not run on hardware
//
#include <hip/hip_runtime.h>
#include <hip/hip_bf16.h>

// ---------------------------------------------------------------------------
// Dilated LSTM (B=8, T=256, C=1024, H=2048, dilation=4) for gfx1250 (MI455X).
//
//   * Dilation 4 => 4 consecutive timesteps independent: scan = 64 sequential
//     GEMMs of (M=32, K=2048, N=8192) + gate nonlinearity.
//   * All GEMMs via v_wmma_f32_16x16x32_bf16 (f32 accumulate).
//   * Phase 2 is a PERSISTENT kernel (128 blocks, device-wide atomic barrier):
//     64 kernel launches @ ~1-2us each would dominate the sub-us per-step
//     compute (w_hh bf16 = 32MB, resident in the 192MB L2).
//   * h (A-matrix, 128KB) is staged into LDS with double-buffered
//     global_load_async_to_lds_b128 (ASYNCcnt) overlapping the WMMA loop;
//     cell state c lives in LDS for all 64 steps.
// ---------------------------------------------------------------------------

#define Bsz   8
#define Tsz   256
#define Csz   1024
#define Hsz   2048
#define G4H   8192      // 4*H
#define MROWS 32        // dilation * B
#define NBLK  128       // persistent grid size (Hsz/16 column tiles)
#define AK    256       // A-chunk K width staged in LDS
#define NCHUNK (Hsz / AK)

typedef __attribute__((ext_vector_type(16))) __bf16 bf16x16;
typedef __attribute__((ext_vector_type(8)))  __bf16 bf16x8;
typedef __attribute__((ext_vector_type(8)))  float  f32x8;
typedef __attribute__((ext_vector_type(4)))  int    v4i;

// ---- optional CDNA5 async global->LDS path --------------------------------
#if defined(__has_builtin)
#if __has_builtin(__builtin_amdgcn_global_load_async_to_lds_b128) && \
    __has_builtin(__builtin_amdgcn_s_wait_asynccnt)
#define HAVE_ASYNC_LDS 1
#endif
#endif

typedef __attribute__((address_space(1))) v4i* gv4i_p;   // global int4*
typedef __attribute__((address_space(3))) v4i* lv4i_p;   // LDS int4*

__device__ __forceinline__ void async_copy_b128(const __bf16* gsrc, __bf16* ldst) {
#ifdef HAVE_ASYNC_LDS
    __builtin_amdgcn_global_load_async_to_lds_b128((gv4i_p)gsrc, (lv4i_p)ldst, 0, 0);
#else
    *(bf16x8*)ldst = *(const bf16x8*)gsrc;   // fallback: sync load + ds store
#endif
}
__device__ __forceinline__ void wait_async_all() {
#ifdef HAVE_ASYNC_LDS
    __builtin_amdgcn_s_wait_asynccnt(0);
#endif
}

// ---- WMMA fragment loader (generic pointer: works for global and LDS) -----
// 16-bit A(16x32)/B(32x16) fragment, CDNA5 layout: lane l -> row/col l%16,
// K-offset (l<16?0:8); elems 0..7 = K[koff..koff+7], 8..15 = K[koff+16..+23].
__device__ __forceinline__ bf16x16 load_frag16x32(const __bf16* p0) {
    bf16x8 lo = *(const bf16x8*)(p0);
    bf16x8 hi = *(const bf16x8*)(p0 + 16);
    bf16x16 r;
#pragma unroll
    for (int i = 0; i < 8; ++i) { r[i] = lo[i]; r[8 + i] = hi[i]; }
    return r;
}

__device__ __forceinline__ float sigf(float x) {
    return 1.0f / (1.0f + __expf(-x));
}
__device__ __forceinline__ float tanh_fast(float x) {
    x = fminf(fmaxf(x, -15.0f), 15.0f);
    float e = __expf(2.0f * x);
    return (e - 1.0f) / (e + 1.0f);
}

// ---- device-wide barrier for the persistent kernel ------------------------
__device__ __forceinline__ void grid_sync(unsigned* cnt, unsigned* gen, unsigned nblocks) {
    __syncthreads();
    if (threadIdx.x == 0) {
        __threadfence();
        unsigned g = __hip_atomic_load(gen, __ATOMIC_RELAXED, __HIP_MEMORY_SCOPE_AGENT);
        unsigned a = __hip_atomic_fetch_add(cnt, 1u, __ATOMIC_ACQ_REL, __HIP_MEMORY_SCOPE_AGENT);
        if (a == nblocks - 1u) {
            __hip_atomic_store(cnt, 0u, __ATOMIC_RELAXED, __HIP_MEMORY_SCOPE_AGENT);
            __hip_atomic_store(gen, g + 1u, __ATOMIC_RELEASE, __HIP_MEMORY_SCOPE_AGENT);
        } else {
            while (__hip_atomic_load(gen, __ATOMIC_ACQUIRE, __HIP_MEMORY_SCOPE_AGENT) == g)
                __builtin_amdgcn_s_sleep(1);
        }
    }
    __syncthreads();
}

// ---- conversion kernels ----------------------------------------------------
__global__ __launch_bounds__(256) void cvt_f32_to_bf16(const float* __restrict__ src,
                                                       __bf16* __restrict__ dst, int n) {
    int i = blockIdx.x * 256 + threadIdx.x;
    if (i < n) dst[i] = (__bf16)src[i];
}

// x: (B,T,C) f32 -> xt: (T*B, C) bf16
__global__ __launch_bounds__(256) void cvt_x_transpose(const float* __restrict__ x,
                                                       __bf16* __restrict__ xt) {
    int i = blockIdx.x * 256 + threadIdx.x;
    if (i >= Bsz * Tsz * Csz) return;
    int cpos = i % Csz;
    int bt   = i / Csz;
    int b = bt / Tsz, t = bt % Tsz;
    xt[((size_t)t * Bsz + b) * (size_t)Csz + cpos] = (__bf16)x[i];
}

// ---- phase 1: u = xt @ w_ih^T + (b_ih + b_hh) ------------------------------
__global__ __launch_bounds__(256)
void ih_gemm_kernel(const __bf16* __restrict__ xt,    // (T*B, C)
                    const __bf16* __restrict__ wih,   // (4H, C)
                    const float*  __restrict__ b_ih,
                    const float*  __restrict__ b_hh,
                    float* __restrict__ u)            // (T*B, 4H)
{
    const int lane  = threadIdx.x & 31;
    const int wave  = threadIdx.x >> 5;
    const int mt    = wave & 1;
    const int nt    = wave >> 1;
    const int mbase = blockIdx.y * 32 + mt * 16;
    const int nbase = blockIdx.x * 64 + nt * 16;
    const int lr    = lane & 15;
    const int half  = lane >> 4;
    const int koff0 = half * 8;

    const __bf16* pa = xt  + (size_t)(mbase + lr) * Csz + koff0;
    const __bf16* pb = wih + (size_t)(nbase + lr) * Csz + koff0;

    f32x8 acc = {};
#pragma unroll 4
    for (int k = 0; k < Csz; k += 32) {
        bf16x16 a = load_frag16x32(pa + k);
        bf16x16 b = load_frag16x32(pb + k);
        acc = __builtin_amdgcn_wmma_f32_16x16x32_bf16(
            false, a, false, b, (short)0, acc, false, false);
    }
    const int   col  = nbase + lr;
    const float bias = b_ih[col] + b_hh[col];
#pragma unroll
    for (int v = 0; v < 8; ++v) {
        int row = mbase + v + half * 8;
        u[(size_t)row * G4H + col] = acc[v] + bias;
    }
}

// ---- phase 2: persistent recurrent kernel ----------------------------------
// 128 blocks x 256 threads. Block owns 16 hidden columns for ALL 64 steps:
// waves = 2 m-tiles x 4 gates; c lives in LDS; h ping-pongs through global.
__global__ __launch_bounds__(256)
void lstm_scan_kernel(const float*  __restrict__ u,     // (T*B, 4H), biases folded
                      const __bf16* __restrict__ whh,   // (4H, H) bf16
                      __bf16*       __restrict__ hbuf,  // (2, 32, H) bf16 ping-pong
                      float*        __restrict__ out,   // (B, T, H) f32
                      unsigned*     __restrict__ sync)  // [cnt, gen]
{
    __shared__ __bf16 abuf[2][MROWS][AK];   // 2 x 16KB staged A chunks
    __shared__ float  xch[4][MROWS][16];    // gate exchange, 8KB
    __shared__ float  clds[MROWS][16];      // cell state, 2KB

    const int lane  = threadIdx.x & 31;
    const int wave  = threadIdx.x >> 5;
    const int mt    = wave & 1;
    const int gate  = wave >> 1;            // 0=i 1=f 2=g 3=o
    const int n0    = blockIdx.x * 16;
    const int lr    = lane & 15;
    const int half  = lane >> 4;
    const int koff0 = half * 8;

    // zero cell state
    for (int idx = threadIdx.x; idx < MROWS * 16; idx += 256)
        clds[idx >> 4][idx & 15] = 0.0f;
    __syncthreads();

    const __bf16* pb = whh + ((size_t)gate * Hsz + n0 + lr) * (size_t)Hsz + koff0;

    for (int g = 0; g < Tsz / 4; ++g) {
        const __bf16* h_in  = hbuf + (size_t)(g & 1) * MROWS * Hsz;
        __bf16*       h_out = hbuf + (size_t)((g + 1) & 1) * MROWS * Hsz;

        // --- stage chunk 0 of A (32 x AK bf16) into LDS ---
        // granule = 16B; 32*AK*2/16 = 1024 granules over 256 threads
#pragma unroll
        for (int i = 0; i < 4; ++i) {
            int gran = threadIdx.x * 4 + i;
            int row  = gran >> 5;           // AK/8 = 32 granules per row
            int k16  = gran & 31;
            async_copy_b128(h_in + (size_t)row * Hsz + k16 * 8,
                            &abuf[0][row][k16 * 8]);
        }
        wait_async_all();
        __syncthreads();

        f32x8 acc = {};
        for (int kc = 0; kc < NCHUNK; ++kc) {
            const int buf = kc & 1;
            // issue next chunk into the other buffer (overlaps with WMMAs)
            if (kc + 1 < NCHUNK) {
#pragma unroll
                for (int i = 0; i < 4; ++i) {
                    int gran = threadIdx.x * 4 + i;
                    int row  = gran >> 5;
                    int k16  = gran & 31;
                    async_copy_b128(h_in + (size_t)row * Hsz + (kc + 1) * AK + k16 * 8,
                                    &abuf[buf ^ 1][row][k16 * 8]);
                }
            }
            // --- WMMA over this chunk: A from LDS, B (weights) from L2 ---
            const __bf16* pa = &abuf[buf][mt * 16 + lr][koff0];
#pragma unroll
            for (int kk = 0; kk < AK; kk += 32) {
                __builtin_prefetch(pb + (size_t)kc * AK + kk + 256, 0, 1);
                bf16x16 a = load_frag16x32(pa + kk);
                bf16x16 b = load_frag16x32(pb + (size_t)kc * AK + kk);
                acc = __builtin_amdgcn_wmma_f32_16x16x32_bf16(
                    false, a, false, b, (short)0, acc, false, false);
            }
            wait_async_all();     // own next-chunk loads landed
            __syncthreads();      // everyone done reading buf / writing buf^1
        }

        // --- gate exchange ---
#pragma unroll
        for (int v = 0; v < 8; ++v)
            xch[gate][mt * 16 + v + half * 8][lr] = acc[v];
        __syncthreads();

        // --- cell update: 32x16 elements over 256 threads (stable mapping) ---
        for (int idx = threadIdx.x; idx < MROWS * 16; idx += 256) {
            int m  = idx >> 4;
            int n  = idx & 15;
            int nh = n0 + n;
            size_t urow = (size_t)(32 * g + m) * G4H;
            float iv = xch[0][m][n] + u[urow + nh];
            float fv = xch[1][m][n] + u[urow + Hsz + nh];
            float gv = xch[2][m][n] + u[urow + 2 * Hsz + nh];
            float ov = xch[3][m][n] + u[urow + 3 * Hsz + nh];

            float c_old = clds[m][n];
            float c_new = sigf(fv) * c_old + sigf(iv) * tanh_fast(gv);
            float h_new = sigf(ov) * tanh_fast(c_new);
            clds[m][n] = c_new;

            h_out[(size_t)m * Hsz + nh] = (__bf16)h_new;
            int j = m >> 3, b = m & 7;
            int t = 4 * g + j;
            out[((size_t)b * Tsz + t) * (size_t)Hsz + nh] = h_new;
        }

        grid_sync(&sync[0], &sync[1], NBLK);
    }
}

// ---------------------------------------------------------------------------
extern "C" void kernel_launch(void* const* d_in, const int* in_sizes, int n_in,
                              void* d_out, int out_size, void* d_ws, size_t ws_size,
                              hipStream_t stream) {
    const float* x    = (const float*)d_in[0];   // (B,T,C)
    const float* w_ih = (const float*)d_in[1];   // (4H,C)
    const float* b_ih = (const float*)d_in[2];   // (4H)
    const float* w_hh = (const float*)d_in[3];   // (4H,H)
    const float* b_hh = (const float*)d_in[4];   // (4H)
    float* out = (float*)d_out;                  // (B,T,H)

    // --- workspace layout ---
    char* ws = (char*)d_ws;
    size_t off = 0;
    float*  u    = (float*)(ws + off);    off += (size_t)Tsz * Bsz * G4H * 4;   // 64 MB
    __bf16* xt   = (__bf16*)(ws + off);   off += (size_t)Tsz * Bsz * Csz * 2;   //  4 MB
    __bf16* wihb = (__bf16*)(ws + off);   off += (size_t)G4H * Csz * 2;         // 16 MB
    __bf16* whhb = (__bf16*)(ws + off);   off += (size_t)G4H * Hsz * 2;         // 32 MB
    __bf16* hbuf = (__bf16*)(ws + off);   off += (size_t)2 * MROWS * Hsz * 2;   // 256 KB
    unsigned* syncw = (unsigned*)(ws + off); off += 256;                        // barrier state
    if (ws_size < off) return;

    // --- conversions to bf16 ---
    {
        int n = Bsz * Tsz * Csz;
        cvt_x_transpose<<<(n + 255) / 256, 256, 0, stream>>>(x, xt);
    }
    {
        int n = G4H * Csz;
        cvt_f32_to_bf16<<<(n + 255) / 256, 256, 0, stream>>>(w_ih, wihb, n);
    }
    {
        int n = G4H * Hsz;
        cvt_f32_to_bf16<<<(n + 255) / 256, 256, 0, stream>>>(w_hh, whhb, n);
    }

    // --- phase 1: input-projection GEMM, biases (b_ih + b_hh) folded in ---
    ih_gemm_kernel<<<dim3(G4H / 64, (Tsz * Bsz) / 32), 256, 0, stream>>>(
        xt, wihb, b_ih, b_hh, u);

    // --- init: h ping buffer (parity 0) = 0, barrier counters = 0 ---
    (void)hipMemsetAsync(hbuf, 0, (size_t)MROWS * Hsz * 2, stream);
    (void)hipMemsetAsync(syncw, 0, 256, stream);

    // --- phase 2: single persistent kernel runs all 64 dilation-groups ---
    lstm_scan_kernel<<<NBLK, 256, 0, stream>>>(u, whhb, hbuf, out, syncw);
}
